// Transformers__86698209837724
// MI455X (gfx1250) — compile-verified, matrix operand order checked by reading, not run on hardware
//
#include <hip/hip_runtime.h>
#include <hip/hip_bf16.h>

#define NODES  50000
#define GRAPHS 250
#define EDGES  1600000
#define INCH   200
#define HID    128

// ---- plain ext-vector types (trivial, union-safe) ----
typedef __attribute__((ext_vector_type(16))) __bf16    v16bf;
typedef __attribute__((ext_vector_type(8)))  float     v8f;
typedef __attribute__((ext_vector_type(4)))  unsigned  uint4e;
typedef __attribute__((ext_vector_type(2)))  unsigned  uint2e;
typedef __attribute__((ext_vector_type(4)))  float     float4e;

union FragAB {
    unsigned u[8];
    uint4e   q[2];
    v16bf    v;
};

__device__ __forceinline__ unsigned pack_bf16(float lo, float hi) {
    // round-to-nearest-even f32 -> bf16, packed pair
    unsigned a = __float_as_uint(lo);
    unsigned b = __float_as_uint(hi);
    unsigned ra = (a + 0x7FFFu + ((a >> 16) & 1u)) >> 16;
    unsigned rb = (b + 0x7FFFu + ((b >> 16) & 1u)) >> 16;
    return (ra & 0xFFFFu) | (rb << 16);
}

// -----------------------------------------------------------------------------
// Repack W[k][n] (f32, K x 128) into bf16 B-fragments, WMMA 16x16x32 layout:
//   wfrag[chunk][tile(8)][lane(32)][vgpr(8)] : uint = bf16 pair (k, k+1)
//   lane n = (lane&15)+16*tile ; kpair = (j&3) + ((j>>2)<<3) + ((lane>>4)<<2)
// Zero-padded for k >= K (layer 1: K=200, chunks=7 -> Kpad=224).
// -----------------------------------------------------------------------------
__global__ void w_pack_kernel(const float* __restrict__ W, unsigned* __restrict__ wfrag,
                              int K, int chunks) {
    int idx = blockIdx.x * blockDim.x + threadIdx.x;
    int total = chunks * 2048;
    if (idx >= total) return;
    int ch   = idx >> 11;
    int rem  = idx & 2047;
    int t    = rem >> 8;
    int lane = (rem >> 3) & 31;
    int j    = rem & 7;
    int n    = (lane & 15) + 16 * t;
    int kp   = (j & 3) + ((j >> 2) << 3) + ((lane >> 4) << 2);
    int k    = ch * 32 + 2 * kp;
    unsigned v = 0u;
    if (k < K)  // K is even, so k+1 < K too
        v = pack_bf16(W[(size_t)k * HID + n], W[(size_t)(k + 1) * HID + n]);
    wfrag[idx] = v;
}

// -----------------------------------------------------------------------------
// Edge scatter-add: agg[dst[e], c..c+3] += x[src[e], c..c+3]
// grid = (ceil(E/256), K/4). Edge lists (12.8 MB) and feature rows are
// L2-resident on MI455X (192 MB L2); throughput limit is L2 atomics.
// -----------------------------------------------------------------------------
__global__ void scatter_kernel(const float* __restrict__ x, const int* __restrict__ src,
                               const int* __restrict__ dst, float* __restrict__ agg, int K) {
    int e = blockIdx.x * blockDim.x + threadIdx.x;
    if (e >= EDGES) return;
    int c = blockIdx.y * 4;
    int s = src[e];
    int d = dst[e];
    float4e v = *(const float4e*)(x + (size_t)s * K + c);
    float* a = agg + (size_t)d * K + c;
    atomicAdd(a + 0, v.x);
    atomicAdd(a + 1, v.y);
    atomicAdd(a + 2, v.z);
    atomicAdd(a + 3, v.w);
}

// -----------------------------------------------------------------------------
// out[row, 0..127] = relu( (X[row,:] + AGG[row,:]) @ W + b )
// 256 threads = 8 waves; block does 128 rows x 128 cols.
// Per 32-K chunk:
//   * stage packed-bf16 (X+AGG) tile into LDS (float4 global loads, b64 DS
//     stores; 16B-aligned rows: stride 20 dwords)
//   * each lane assembles its A fragment with two ds_load_b128
//   * B fragments stream via an explicit 2-stage software pipeline
//     (bcur/bnext ping-pong) so the next tile's global_load_b128 clause
//     overlaps the current v_wmma (wait loadcnt<=2, not 0).
// -----------------------------------------------------------------------------
__global__ void __launch_bounds__(256) gin_gemm_kernel(
    const float* __restrict__ X, const float* __restrict__ AGG,
    const unsigned* __restrict__ wfrag, const float* __restrict__ bias,
    float* __restrict__ out, int K, int chunks)
{
    __shared__ unsigned sA[128][20];   // [row][kpair 0..15], stride 20 -> 80B rows (16B aligned)
    const int tid  = threadIdx.x;
    const int lane = tid & 31;
    const int wave = tid >> 5;
    const int half = lane >> 4;
    const int blockRow = blockIdx.x * 128;

    v8f acc[8] = {};

    for (int ch = 0; ch < chunks; ++ch) {
        __syncthreads();  // protect previous iteration's sA reads
        #pragma unroll
        for (int i = 0; i < 4; ++i) {
            int idx = i * 256 + tid;      // 1024 = 128 rows * 8 quads
            int r   = idx >> 3;
            int q   = idx & 7;
            int row = blockRow + r;
            int k   = ch * 32 + q * 4;
            uint2e v2 = (uint2e)0u;
            if (row < NODES && k < K) {   // K % 4 == 0, so k..k+3 all valid
                float4e xv = *(const float4e*)(X   + (size_t)row * K + k);
                float4e av = *(const float4e*)(AGG + (size_t)row * K + k);
                v2.x = pack_bf16(xv.x + av.x, xv.y + av.y);
                v2.y = pack_bf16(xv.z + av.z, xv.w + av.w);
            }
            *(uint2e*)&sA[r][q * 2] = v2;
        }
        __syncthreads();

        // A fragment: u[0..3] = kpairs {0..3}+4*half, u[4..7] = kpairs {8..11}+4*half
        FragAB afr;
        const unsigned* arowp = &sA[wave * 16 + (lane & 15)][0];
        afr.q[0] = *(const uint4e*)(arowp + half * 4);
        afr.q[1] = *(const uint4e*)(arowp + 8 + half * 4);

        const uint4e* bbase = (const uint4e*)(wfrag + (size_t)ch * 2048) + lane * 2;
        if (ch + 1 < chunks)
            __builtin_prefetch(wfrag + (size_t)(ch + 1) * 2048 + lane * 8, 0, 0);

        // 2-stage software pipeline over the 8 N-tiles
        FragAB bcur, bnext;
        bcur.q[0] = bbase[0];
        bcur.q[1] = bbase[1];
        #pragma unroll
        for (int t = 0; t < 8; ++t) {
            if (t + 1 < 8) {
                bnext.q[0] = bbase[(t + 1) * 64 + 0];
                bnext.q[1] = bbase[(t + 1) * 64 + 1];
            }
            acc[t] = __builtin_amdgcn_wmma_f32_16x16x32_bf16(
                         false, afr.v, false, bcur.v, (short)0, acc[t], false, false);
            bcur = bnext;
        }
    }

    // Epilogue: C layout lane(0..15)=N, VGPR v -> M = v + 8*half
    #pragma unroll
    for (int t = 0; t < 8; ++t) {
        int col = t * 16 + (lane & 15);
        float bv = bias[col];
        #pragma unroll
        for (int m = 0; m < 8; ++m) {
            int row = blockRow + wave * 16 + m + half * 8;
            if (row < NODES) {
                float val = acc[t][m] + bv;
                out[(size_t)row * HID + col] = val > 0.f ? val : 0.f;
            }
        }
    }
}

// -----------------------------------------------------------------------------
__global__ void pool_kernel(const float* __restrict__ h, const int* __restrict__ batch,
                            float* __restrict__ gsum) {
    int i = blockIdx.x * blockDim.x + threadIdx.x;
    if (i >= NODES) return;
    int c = blockIdx.y * 4;
    int g = batch[i];
    float4e v = *(const float4e*)(h + (size_t)i * HID + c);
    float* a = gsum + (size_t)g * HID + c;
    atomicAdd(a + 0, v.x);
    atomicAdd(a + 1, v.y);
    atomicAdd(a + 2, v.z);
    atomicAdd(a + 3, v.w);
}

__global__ void count_kernel(const int* __restrict__ batch, float* __restrict__ cnt) {
    int i = blockIdx.x * blockDim.x + threadIdx.x;
    if (i >= NODES) return;
    atomicAdd(&cnt[batch[i]], 1.0f);
}

__global__ void head_kernel(const float* __restrict__ gfc, const float* __restrict__ gsc,
                            const float* __restrict__ cnt, const float* __restrict__ Wp,
                            const float* __restrict__ bp, float* __restrict__ out) {
    int idx = blockIdx.x * blockDim.x + threadIdx.x;
    if (idx >= GRAPHS * 2) return;
    int g = idx >> 1;
    int v = idx & 1;
    float c = cnt[g];
    c = c > 1.f ? c : 1.f;
    float a = bp[v];
    #pragma unroll 8
    for (int k = 0; k < HID; ++k)
        a += ((gfc[g * HID + k] + gsc[g * HID + k]) / c) * Wp[k * 2 + v];
    out[idx] = 1.f / (1.f + __expf(-a));
}

// -----------------------------------------------------------------------------
extern "C" void kernel_launch(void* const* d_in, const int* in_sizes, int n_in,
                              void* d_out, int out_size, void* d_ws, size_t ws_size,
                              hipStream_t stream) {
    const float* x_fc    = (const float*)d_in[0];
    const int*   edge_fc = (const int*)  d_in[1];
    const float* x_sc    = (const float*)d_in[2];
    const int*   edge_sc = (const int*)  d_in[3];
    const int*   batch   = (const int*)  d_in[4];
    const float* W1 = (const float*)d_in[5];
    const float* b1 = (const float*)d_in[6];
    const float* W2 = (const float*)d_in[7];
    const float* b2 = (const float*)d_in[8];
    const float* W3 = (const float*)d_in[9];
    const float* b3 = (const float*)d_in[10];
    const float* Wp = (const float*)d_in[11];
    const float* bp = (const float*)d_in[12];

    char* ws = (char*)d_ws;
    auto alloc = [&](size_t bytes) -> char* {
        char* p = ws;
        ws += (bytes + 255) & ~(size_t)255;
        return p;
    };
    float*    agg = (float*)   alloc((size_t)NODES * INCH * 4);  // max-K agg buffer
    float*    hA  = (float*)   alloc((size_t)NODES * HID * 4);
    float*    hB  = (float*)   alloc((size_t)NODES * HID * 4);
    float*    gfc = (float*)   alloc((size_t)GRAPHS * HID * 4);
    float*    gsc = (float*)   alloc((size_t)GRAPHS * HID * 4);
    float*    cnt = (float*)   alloc((size_t)GRAPHS * 4);
    unsigned* wf1 = (unsigned*)alloc((size_t)7 * 2048 * 4);
    unsigned* wf2 = (unsigned*)alloc((size_t)4 * 2048 * 4);
    unsigned* wf3 = (unsigned*)alloc((size_t)4 * 2048 * 4);

    dim3 b256(256);
    // weight repack (cheap, rerun every launch for determinism)
    w_pack_kernel<<<(7 * 2048 + 255) / 256, b256, 0, stream>>>(W1, wf1, INCH, 7);
    w_pack_kernel<<<(4 * 2048 + 255) / 256, b256, 0, stream>>>(W2, wf2, HID, 4);
    w_pack_kernel<<<(4 * 2048 + 255) / 256, b256, 0, stream>>>(W3, wf3, HID, 4);

    hipMemsetAsync(cnt, 0, (size_t)GRAPHS * 4, stream);
    hipMemsetAsync(gfc, 0, (size_t)GRAPHS * HID * 4, stream);
    hipMemsetAsync(gsc, 0, (size_t)GRAPHS * HID * 4, stream);
    count_kernel<<<(NODES + 255) / 256, b256, 0, stream>>>(batch, cnt);

    const int gemmGrid = (NODES + 127) / 128;
    const int edgeGrid = (EDGES + 255) / 256;

    auto run_branch = [&](const float* x, const int* edge, float* gsum) {
        const int* src = edge;
        const int* dst = edge + EDGES;
        // Layer 1 (K = 200, 7 chunks of 32 with zero pad)
        hipMemsetAsync(agg, 0, (size_t)NODES * INCH * 4, stream);
        scatter_kernel<<<dim3(edgeGrid, INCH / 4), b256, 0, stream>>>(x, src, dst, agg, INCH);
        gin_gemm_kernel<<<gemmGrid, b256, 0, stream>>>(x, agg, wf1, b1, hA, INCH, 7);
        // Layer 2 (K = 128)
        hipMemsetAsync(agg, 0, (size_t)NODES * HID * 4, stream);
        scatter_kernel<<<dim3(edgeGrid, HID / 4), b256, 0, stream>>>(hA, src, dst, agg, HID);
        gin_gemm_kernel<<<gemmGrid, b256, 0, stream>>>(hA, agg, wf2, b2, hB, HID, 4);
        // Layer 3 (K = 128)
        hipMemsetAsync(agg, 0, (size_t)NODES * HID * 4, stream);
        scatter_kernel<<<dim3(edgeGrid, HID / 4), b256, 0, stream>>>(hB, src, dst, agg, HID);
        gin_gemm_kernel<<<gemmGrid, b256, 0, stream>>>(hB, agg, wf3, b3, hA, HID, 4);
        // mean-pool sums
        pool_kernel<<<dim3((NODES + 255) / 256, HID / 4), b256, 0, stream>>>(hA, batch, gsum);
    };
    run_branch(x_fc, edge_fc, gfc);
    run_branch(x_sc, edge_sc, gsc);

    head_kernel<<<(GRAPHS * 2 + 255) / 256, b256, 0, stream>>>(gfc, gsc, cnt, Wp, bp,
                                                               (float*)d_out);
}
